// FusionMamba_33311766347949
// MI455X (gfx1250) — matrix operand order repaired and checked by reading, not compiled, not run on hardware
//
// FusionMamba for MI455X (gfx1250, wave32, WMMA).
//
// Design (see analysis): all dense projections as bf16 WMMA (v_wmma_f32_16x16x32_bf16,
// f32 accumulate), selective scan as a 64-chunk 3-pass parallel scan (4096 waves),
// wave-per-token LayerNorms with __shfl_xor reductions. Whole working set (~80MB)
// stays L2-resident (192MB); HBM traffic is just inputs+output.
//
// Param flattening assumption: top-level in setup_inputs() insertion order
// (img1, img2, img1_sum, img2_sum, params); `params` flattened jax-pytree style
// (dicts sorted by key, lists in order):
//   4 alpha, 5 beta, 6 op_W1, 7 op_W2, 8 op_b1, 9 op_b2, 10 op_ln_b, 11 op_ln_g,
//   12..33 self_attn, 34..66 spa[0], 67..99 spa[1], 100..132 spa_cross,
//   133..165 spe[0], 166..198 spe[1], 199..231 spe_cross.
// Cross block (33 leaves): W_cw,b_cw,cwn_b,cwn_g, [mamba: W_in,W_in_e,W_out,
//   bwd{A_log,D,W_dt,W_x,b_conv,b_conv_e,b_dt,w_conv,w_conv_e}, fwd{same}, ln_b,ln_g],
//   n0_b,n0_g,n1_b,n1_g,pn_b,pn_g.
// Single block (22 leaves): [mamba: W_in,W_out, bwd{A_log,D,W_dt,W_x,b_conv,b_dt,w_conv},
//   fwd{same}, ln_b,ln_g], n_b,n_g,pn_b,pn_g.

#include <hip/hip_runtime.h>

#define LSEQ 16384        // H*W
#define DMODEL 64
#define NSTATE 16
#define NCHUNK 64
#define CHUNK  256

typedef __attribute__((ext_vector_type(8)))  __bf16 v8bf;
typedef __attribute__((ext_vector_type(16))) __bf16 v16bf;
typedef __attribute__((ext_vector_type(8)))  float  v8f;

// ---------------- device helpers ----------------
__device__ __forceinline__ float siluf(float x)     { return x / (1.f + expf(-x)); }
__device__ __forceinline__ float softplusf(float x) { return x > 20.f ? x : log1pf(expf(x)); }
__device__ __forceinline__ float geluf(float x)     { return 0.5f * x * (1.f + erff(x * 0.70710678118654752f)); }
__device__ __forceinline__ float sigclipf(float x)  {
  float s = 1.f / (1.f + expf(-x));
  return fminf(fmaxf(s, 0.01f), 0.99f);
}
__device__ __forceinline__ float wave_sum64(float s) {
  #pragma unroll
  for (int o = 16; o; o >>= 1) s += __shfl_xor(s, o, 32);
  return s;
}

// ---------------- layout shuffles ----------------
__global__ __launch_bounds__(256) void k_seq_from_img(const float* __restrict__ img,
                                                      float* __restrict__ seq) {
  int t = blockIdx.x * 256 + threadIdx.x;          // 64*LSEQ threads
  int c = t >> 14, l = t & (LSEQ - 1);
  seq[(size_t)l * DMODEL + c] = img[t];
}
__global__ __launch_bounds__(256) void k_seq_to_img(const float* __restrict__ seq,
                                                    float* __restrict__ img) {
  int t = blockIdx.x * 256 + threadIdx.x;
  int c = t >> 14, l = t & (LSEQ - 1);
  img[t] = seq[(size_t)l * DMODEL + c];
}

// ---------------- weight -> padded bf16 ----------------
__global__ void k_w2bf(const float* __restrict__ w, int rows, int cols, int colpad,
                       int total, __bf16* __restrict__ o) {
  int t = blockIdx.x * blockDim.x + threadIdx.x;
  if (t >= total) return;
  int r = t / colpad, k = t - r * colpad;
  float v = (r < rows && k < cols) ? w[r * cols + k] : 0.f;
  o[t] = (__bf16)v;
}

// ---------------- WMMA GEMM: Y = X * W^T (+bias, epilogue) ----------------
// X: (LSEQ x K) bf16 row stride lda; W: (Npad x K) bf16 row stride ldw.
// One wave -> one 16x16 tile. EPI: 0 none, 1 softplus, 2 gelu.
template <int EPI>
__global__ __launch_bounds__(256) void k_gemm(const __bf16* __restrict__ A, int lda,
                                              const __bf16* __restrict__ W, int ldw,
                                              const float* __restrict__ bias,
                                              float* __restrict__ Cf, int ldc,
                                              __bf16* __restrict__ Cb, int ldcb,
                                              int Kdim) {
  const int lane = threadIdx.x & 31;
  const int wv   = threadIdx.x >> 5;
  const int mt   = blockIdx.x * 8 + wv;   // M-tile (1024 total)
  const int nt   = blockIdx.y;            // N-tile
  const int row  = lane & 15;
  const int hi   = lane >> 4;

  const __bf16* ap = A + (size_t)(mt * 16 + row) * lda + hi * 8;
  const __bf16* bp = W + (size_t)(nt * 16 + row) * ldw + hi * 8;

  v8f acc = {0.f, 0.f, 0.f, 0.f, 0.f, 0.f, 0.f, 0.f};
  for (int kk = 0; kk < Kdim; kk += 32) {
    __builtin_prefetch(ap + kk + 32, 0, 1);   // -> global_prefetch_b8
    v8bf a0 = *(const v8bf*)(ap + kk);
    v8bf a1 = *(const v8bf*)(ap + kk + 16);
    v8bf b0 = *(const v8bf*)(bp + kk);
    v8bf b1 = *(const v8bf*)(bp + kk + 16);
    v16bf av = __builtin_shufflevector(a0, a1, 0,1,2,3,4,5,6,7,8,9,10,11,12,13,14,15);
    v16bf bv = __builtin_shufflevector(b0, b1, 0,1,2,3,4,5,6,7,8,9,10,11,12,13,14,15);
    acc = __builtin_amdgcn_wmma_f32_16x16x32_bf16(false, av, false, bv,
                                                  (short)0, acc, false, false);
  }
  const int n  = nt * 16 + row;
  const float bval = bias ? bias[n] : 0.f;
  #pragma unroll
  for (int r = 0; r < 8; ++r) {
    int m = mt * 16 + r + hi * 8;          // C/D layout: VGPR r -> M = r + 8*hi
    float v = acc[r] + bval;
    if (EPI == 1) v = softplusf(v);
    if (EPI == 2) v = geluf(v);
    Cf[(size_t)m * ldc + n] = v;
    if (Cb) Cb[(size_t)m * ldcb + n] = (__bf16)v;
  }
}

// ---------------- depthwise causal conv (K=4) + silu ----------------
__global__ __launch_bounds__(256) void k_conv_silu(const float* __restrict__ x, int ldx,
                                                   int rev,
                                                   const float* __restrict__ wc,
                                                   const float* __restrict__ bc,
                                                   float* __restrict__ uf,
                                                   __bf16* __restrict__ ub) {
  int t = blockIdx.x * 256 + threadIdx.x;   // LSEQ*64
  int c = t & 63, l = t >> 6;
  float acc = bc[c];
  #pragma unroll
  for (int k = 0; k < 4; ++k) {
    int j = l - 3 + k;
    if (j >= 0) {
      int src = rev ? (LSEQ - 1 - j) : j;
      acc += x[(size_t)src * ldx + c] * wc[c * 4 + k];
    }
  }
  float u = siluf(acc);
  if (uf) uf[t] = u;
  if (ub) ub[t] = (__bf16)u;
}

// ---------------- selective scan: 3-pass chunked parallel scan ----------------
// dbc layout: (LSEQ x 48): [0:4) dt_r, [4:20) B, [20:36) C.
__global__ __launch_bounds__(256) void k_scan_chunk(const float* __restrict__ dt,
                                                    const float* __restrict__ dbc,
                                                    const float* __restrict__ u,
                                                    const float* __restrict__ Alog,
                                                    float* __restrict__ cA,
                                                    float* __restrict__ cB) {
  int lane = threadIdx.x & 31;
  int wid  = (blockIdx.x * 256 + threadIdx.x) >> 5;   // 0..4095
  int d = wid >> 6, ch = wid & 63, n = lane & 15;
  float Ad = -expf(Alog[d * NSTATE + n]);
  float ap = 1.f, bc = 0.f;
  int l0 = ch * CHUNK;
  for (int i = 0; i < CHUNK; ++i) {
    int l = l0 + i;
    float dtl = dt[(size_t)l * DMODEL + d];
    float ul  = u[(size_t)l * DMODEL + d];
    float Bl  = dbc[(size_t)l * 48 + 4 + n];
    float a = expf(dtl * Ad);
    ap *= a;
    bc = a * bc + dtl * Bl * ul;
  }
  if (lane < 16) { cA[wid * NSTATE + n] = ap; cB[wid * NSTATE + n] = bc; }
}

__global__ void k_scan_prefix(const float* __restrict__ cA, const float* __restrict__ cB,
                              float* __restrict__ h0) {
  int t = blockIdx.x * blockDim.x + threadIdx.x;      // 64*16
  if (t >= DMODEL * NSTATE) return;
  int d = t >> 4, n = t & 15;
  float h = 0.f;
  for (int ch = 0; ch < NCHUNK; ++ch) {
    int idx = (d * NCHUNK + ch) * NSTATE + n;
    h0[idx] = h;
    h = cA[idx] * h + cB[idx];
  }
}

__global__ __launch_bounds__(256) void k_scan_apply(const float* __restrict__ dt,
                                                    const float* __restrict__ dbc,
                                                    const float* __restrict__ u,
                                                    const float* __restrict__ Alog,
                                                    const float* __restrict__ Dp,
                                                    const float* __restrict__ h0,
                                                    float* __restrict__ yacc,
                                                    int rev, int accum) {
  int lane = threadIdx.x & 31;
  int wid  = (blockIdx.x * 256 + threadIdx.x) >> 5;
  int d = wid >> 6, ch = wid & 63, n = lane & 15;
  float Ad = -expf(Alog[d * NSTATE + n]);
  float Dd = Dp[d];
  float h  = h0[wid * NSTATE + n];
  int l0 = ch * CHUNK;
  for (int i = 0; i < CHUNK; ++i) {
    int l = l0 + i;
    float dtl = dt[(size_t)l * DMODEL + d];
    float ul  = u[(size_t)l * DMODEL + d];
    float Bl  = dbc[(size_t)l * 48 + 4 + n];
    float Cl  = dbc[(size_t)l * 48 + 20 + n];
    float a = expf(dtl * Ad);
    h = a * h + dtl * Bl * ul;
    float s = h * Cl;
    #pragma unroll
    for (int o = 8; o; o >>= 1) s += __shfl_xor(s, o, 16);
    if (lane == 0) {
      int lo = rev ? (LSEQ - 1 - l) : l;
      float y = s + ul * Dd;
      if (accum) yacc[(size_t)lo * DMODEL + d] += y;
      else       yacc[(size_t)lo * DMODEL + d]  = y;
    }
  }
}

// ---------------- LayerNorm (wave per token, 2 ch/lane). EPI: 0 none, 1 sigclip.
template <int EPI>
__global__ __launch_bounds__(256) void k_ln(const float* __restrict__ x, int ldx,
                                            const float* __restrict__ g,
                                            const float* __restrict__ b,
                                            float* __restrict__ of, int ldo,
                                            __bf16* __restrict__ ob, int ldob, int oboff) {
  const int lane = threadIdx.x & 31;
  const size_t tok = (size_t)((blockIdx.x * 256 + threadIdx.x) >> 5);
  const float* xr = x + tok * (size_t)ldx;
  float v0 = xr[lane], v1 = xr[lane + 32];
  float m  = wave_sum64(v0 + v1) * (1.f / 64.f);
  float d0 = v0 - m, d1 = v1 - m;
  float var = wave_sum64(d0 * d0 + d1 * d1) * (1.f / 64.f);
  float rs  = rsqrtf(var + 1e-5f);
  float y0 = d0 * rs * g[lane] + b[lane];
  float y1 = d1 * rs * g[lane + 32] + b[lane + 32];
  if (EPI == 1) { y0 = sigclipf(y0); y1 = sigclipf(y1); }
  if (of) { float* o = of + tok * (size_t)ldo; o[lane] = y0; o[lane + 32] = y1; }
  if (ob) { __bf16* o = ob + tok * (size_t)ldob + oboff;
            o[lane] = (__bf16)y0; o[lane + 32] = (__bf16)y1; }
}

// y = LN(0.5*(yf+yb)*silu(z)) -> bf16
__global__ __launch_bounds__(256) void k_gate_ln(const float* __restrict__ ya,
                                                 const float* __restrict__ xz,
                                                 const float* __restrict__ g,
                                                 const float* __restrict__ b,
                                                 __bf16* __restrict__ ob) {
  const int lane = threadIdx.x & 31;
  const size_t tok = (size_t)((blockIdx.x * 256 + threadIdx.x) >> 5);
  float z0 = xz[tok * 128 + 64 + lane], z1 = xz[tok * 128 + 96 + lane];
  float v0 = 0.5f * ya[tok * 64 + lane]      * siluf(z0);
  float v1 = 0.5f * ya[tok * 64 + 32 + lane] * siluf(z1);
  float m  = wave_sum64(v0 + v1) * (1.f / 64.f);
  float d0 = v0 - m, d1 = v1 - m;
  float var = wave_sum64(d0 * d0 + d1 * d1) * (1.f / 64.f);
  float rs  = rsqrtf(var + 1e-5f);
  ob[tok * 64 + lane]      = (__bf16)(d0 * rs * g[lane] + b[lane]);
  ob[tok * 64 + 32 + lane] = (__bf16)(d1 * rs * g[lane + 32] + b[lane + 32]);
}

// ---------------- elementwise combines ----------------
__global__ __launch_bounds__(256) void k_cross_out(const float* __restrict__ pn,
                                                   const float* __restrict__ w,
                                                   const float* __restrict__ n0,
                                                   const float* __restrict__ skip,
                                                   float* __restrict__ out) {
  int t = blockIdx.x * 256 + threadIdx.x;
  float wv = w[t];
  out[t] = pn[t] * wv + n0[t] * (1.f - wv) + skip[t];
}
__global__ __launch_bounds__(256) void k_add(const float* __restrict__ a,
                                             float* __restrict__ out) {
  int t = blockIdx.x * 256 + threadIdx.x;
  out[t] += a[t];
}
__global__ __launch_bounds__(256) void k_fuse(const float* __restrict__ alpha,
                                              const float* __restrict__ beta,
                                              const float* __restrict__ spa,
                                              const float* __restrict__ spe,
                                              float* __restrict__ out) {
  int t = blockIdx.x * 256 + threadIdx.x;
  float a = fminf(fmaxf(alpha[0], 0.1f), 2.f);
  float b = fminf(fmaxf(beta[0],  0.1f), 2.f);
  out[t] = (a * spa[t] + b * spe[t]) / (a + b + 1e-6f);
}

// ================= host side =================
struct DirP {
  const float *A_log, *D, *W_dt, *W_x, *b_conv, *b_conv_e, *b_dt, *w_conv, *w_conv_e;
};
struct MambaP {
  const float *W_in, *W_in_e, *W_out, *ln_b, *ln_g;
  DirP fwd, bwd;
};
struct CrossP {
  const float *W_cw, *b_cw, *cwn_b, *cwn_g, *n0_b, *n0_g, *n1_b, *n1_g, *pn_b, *pn_g;
  MambaP m;
};
struct SingleP {
  const float *n_b, *n_g, *pn_b, *pn_g;
  MambaP m;
};

struct WS {
  float *x1, *x2, *s1, *s2, *bspa, *bspe;
  float *n0, *xz, *e, *u, *dbc, *dt, *yacc, *wg, *t0, *mo;
  __bf16 *cat, *srcbf, *dbcbf, *ybf;
  float *cA, *cB, *h0;
  __bf16 *w_in, *w_ine, *w_out, *w_x[2], *w_dt[2], *w_cw;
};

static WS carve(void* ws) {
  char* base = (char*)ws;
  size_t off = 0;
  auto A = [&](size_t bytes) -> void* {
    void* p = base + off;
    off = (off + bytes + 255) & ~(size_t)255;
    return p;
  };
  const size_t LD = (size_t)LSEQ * DMODEL;
  WS w;
  w.x1   = (float*)A(LD * 4);      w.x2   = (float*)A(LD * 4);
  w.s1   = (float*)A(LD * 4);      w.s2   = (float*)A(LD * 4);
  w.bspa = (float*)A(LD * 4);      w.bspe = (float*)A(LD * 4);
  w.n0   = (float*)A(LD * 4);
  w.xz   = (float*)A((size_t)LSEQ * 128 * 4);
  w.e    = (float*)A(LD * 4);
  w.u    = (float*)A(LD * 4);
  w.dbc  = (float*)A((size_t)LSEQ * 48 * 4);
  w.dt   = (float*)A(LD * 4);
  w.yacc = (float*)A(LD * 4);
  w.wg   = (float*)A(LD * 4);
  w.t0   = (float*)A(LD * 4);
  w.mo   = (float*)A(LD * 4);
  w.cat   = (__bf16*)A((size_t)LSEQ * 128 * 2);
  w.srcbf = (__bf16*)A(LD * 2);
  w.dbcbf = (__bf16*)A((size_t)LSEQ * 48 * 2);
  w.ybf   = (__bf16*)A(LD * 2);
  w.cA = (float*)A((size_t)DMODEL * NCHUNK * NSTATE * 4);
  w.cB = (float*)A((size_t)DMODEL * NCHUNK * NSTATE * 4);
  w.h0 = (float*)A((size_t)DMODEL * NCHUNK * NSTATE * 4);
  w.w_in  = (__bf16*)A(128 * 64 * 2);
  w.w_ine = (__bf16*)A(64 * 64 * 2);
  w.w_out = (__bf16*)A(64 * 64 * 2);
  w.w_x[0] = (__bf16*)A(48 * 64 * 2);  w.w_x[1] = (__bf16*)A(48 * 64 * 2);
  w.w_dt[0] = (__bf16*)A(64 * 32 * 2); w.w_dt[1] = (__bf16*)A(64 * 32 * 2);
  w.w_cw = (__bf16*)A(64 * 128 * 2);
  return w;
}

static void w2bf(hipStream_t st, const float* src, int rows, int cols,
                 int rowpad, int colpad, __bf16* dst) {
  int total = rowpad * colpad;
  k_w2bf<<<(total + 255) / 256, 256, 0, st>>>(src, rows, cols, colpad, total, dst);
}

static void gemm(hipStream_t st, int epi, const __bf16* A, int lda,
                 const __bf16* W, int ldw, const float* bias,
                 float* Cf, int ldc, __bf16* Cb, int ldcb, int ntiles, int K) {
  dim3 g(LSEQ / 16 / 8, ntiles);
  switch (epi) {
    case 0: k_gemm<0><<<g, 256, 0, st>>>(A, lda, W, ldw, bias, Cf, ldc, Cb, ldcb, K); break;
    case 1: k_gemm<1><<<g, 256, 0, st>>>(A, lda, W, ldw, bias, Cf, ldc, Cb, ldcb, K); break;
    default: k_gemm<2><<<g, 256, 0, st>>>(A, lda, W, ldw, bias, Cf, ldc, Cb, ldcb, K); break;
  }
}

static void run_mamba(hipStream_t st, const WS& w, const MambaP& m, bool cross,
                      const __bf16* nbf, int nld, const __bf16* ebf, int eld, float* mo) {
  w2bf(st, m.W_in, 128, 64, 128, 64, w.w_in);
  if (cross) w2bf(st, m.W_in_e, 64, 64, 64, 64, w.w_ine);
  w2bf(st, m.W_out, 64, 64, 64, 64, w.w_out);
  w2bf(st, m.fwd.W_x, 36, 64, 48, 64, w.w_x[0]);
  w2bf(st, m.bwd.W_x, 36, 64, 48, 64, w.w_x[1]);
  w2bf(st, m.fwd.W_dt, 64, 4, 64, 32, w.w_dt[0]);
  w2bf(st, m.bwd.W_dt, 64, 4, 64, 32, w.w_dt[1]);

  gemm(st, 0, nbf, nld, w.w_in, 64, nullptr, w.xz, 128, nullptr, 0, 8, 64);   // xz = x @ W_in^T
  if (cross)
    gemm(st, 0, ebf, eld, w.w_ine, 64, nullptr, w.e, 64, nullptr, 0, 4, 64);  // e = n1 @ W_in_e^T

  const DirP* dirs[2] = {&m.fwd, &m.bwd};
  for (int i = 0; i < 2; ++i) {
    const DirP& dp = *dirs[i];
    int rev = i;
    // u = silu(conv(xi)); src from u (v6) or conv_e(e) (cross)
    k_conv_silu<<<4096, 256, 0, st>>>(w.xz, 128, rev, dp.w_conv, dp.b_conv,
                                      w.u, cross ? (__bf16*)nullptr : w.srcbf);
    if (cross)
      k_conv_silu<<<4096, 256, 0, st>>>(w.e, 64, rev, dp.w_conv_e, dp.b_conv_e,
                                        (float*)nullptr, w.srcbf);
    // dbc = src @ W_x^T   (Nout 36 padded to 48)
    gemm(st, 0, w.srcbf, 64, w.w_x[i], 64, nullptr, w.dbc, 48, w.dbcbf, 48, 3, 64);
    // dt = softplus(dt_r @ W_dt^T + b_dt)  (K padded to 32; zero weight cols)
    gemm(st, 1, w.dbcbf, 48, w.w_dt[i], 32, dp.b_dt, w.dt, 64, nullptr, 0, 4, 32);
    // chunked parallel scan
    k_scan_chunk<<<512, 256, 0, st>>>(w.dt, w.dbc, w.u, dp.A_log, w.cA, w.cB);
    k_scan_prefix<<<4, 256, 0, st>>>(w.cA, w.cB, w.h0);
    k_scan_apply<<<512, 256, 0, st>>>(w.dt, w.dbc, w.u, dp.A_log, dp.D, w.h0,
                                      w.yacc, rev, i);
  }
  k_gate_ln<<<2048, 256, 0, st>>>(w.yacc, w.xz, m.ln_g, m.ln_b, w.ybf);
  gemm(st, 0, w.ybf, 64, w.w_out, 64, nullptr, mo, 64, nullptr, 0, 4, 64);
}

static void run_cross(hipStream_t st, const WS& w, const CrossP& p,
                      const float* x0, const float* x1, float* out) {
  // n0 -> f32 + bf16@cat[:,0:64]; n1 -> bf16@cat[:,64:128]
  k_ln<0><<<2048, 256, 0, st>>>(x0, 64, p.n0_g, p.n0_b, w.n0, 64, w.cat, 128, 0);
  k_ln<0><<<2048, 256, 0, st>>>(x1, 64, p.n1_g, p.n1_b, (float*)nullptr, 0, w.cat, 128, 64);
  // gate w = clip(sigmoid(LN(cat @ W_cw^T + b_cw)))
  w2bf(st, p.W_cw, 64, 128, 64, 128, w.w_cw);
  gemm(st, 0, w.cat, 128, w.w_cw, 128, p.b_cw, w.t0, 64, nullptr, 0, 4, 128);
  k_ln<1><<<2048, 256, 0, st>>>(w.t0, 64, p.cwn_g, p.cwn_b, w.wg, 64, (__bf16*)nullptr, 0, 0);
  // cross mamba: u from n0, dt/B/C from n1
  run_mamba(st, w, p.m, true, w.cat, 128, w.cat + 64, 128, w.mo);
  k_ln<0><<<2048, 256, 0, st>>>(w.mo, 64, p.pn_g, p.pn_b, w.mo, 64, (__bf16*)nullptr, 0, 0);
  k_cross_out<<<4096, 256, 0, st>>>(w.mo, w.wg, w.n0, x0, out);
}

static DirP getDir(void* const* din, int base, bool cross) {
  auto F = [&](int i) { return (const float*)din[base + i]; };
  DirP d;
  if (cross) {
    d.A_log = F(0); d.D = F(1); d.W_dt = F(2); d.W_x = F(3);
    d.b_conv = F(4); d.b_conv_e = F(5); d.b_dt = F(6);
    d.w_conv = F(7); d.w_conv_e = F(8);
  } else {
    d.A_log = F(0); d.D = F(1); d.W_dt = F(2); d.W_x = F(3);
    d.b_conv = F(4); d.b_conv_e = nullptr; d.b_dt = F(5);
    d.w_conv = F(6); d.w_conv_e = nullptr;
  }
  return d;
}
static CrossP getCross(void* const* din, int base) {
  auto F = [&](int i) { return (const float*)din[base + i]; };
  CrossP c;
  c.W_cw = F(0); c.b_cw = F(1); c.cwn_b = F(2); c.cwn_g = F(3);
  c.m.W_in = F(4); c.m.W_in_e = F(5); c.m.W_out = F(6);
  c.m.bwd = getDir(din, base + 7, true);
  c.m.fwd = getDir(din, base + 16, true);
  c.m.ln_b = F(25); c.m.ln_g = F(26);
  c.n0_b = F(27); c.n0_g = F(28); c.n1_b = F(29); c.n1_g = F(30);
  c.pn_b = F(31); c.pn_g = F(32);
  return c;
}
static SingleP getSingle(void* const* din, int base) {
  auto F = [&](int i) { return (const float*)din[base + i]; };
  SingleP s;
  s.m.W_in = F(0); s.m.W_in_e = nullptr; s.m.W_out = F(1);
  s.m.bwd = getDir(din, base + 2, false);
  s.m.fwd = getDir(din, base + 9, false);
  s.m.ln_b = F(16); s.m.ln_g = F(17);
  s.n_b = F(18); s.n_g = F(19); s.pn_b = F(20); s.pn_g = F(21);
  return s;
}

extern "C" void kernel_launch(void* const* d_in, const int* in_sizes, int n_in,
                              void* d_out, int out_size, void* d_ws, size_t ws_size,
                              hipStream_t stream) {
  (void)in_sizes; (void)n_in; (void)out_size; (void)ws_size;
  WS w = carve(d_ws);

  // images -> (L, C) sequences
  k_seq_from_img<<<4096, 256, 0, stream>>>((const float*)d_in[0], w.x1);
  k_seq_from_img<<<4096, 256, 0, stream>>>((const float*)d_in[1], w.x2);
  k_seq_from_img<<<4096, 256, 0, stream>>>((const float*)d_in[2], w.s1);
  k_seq_from_img<<<4096, 256, 0, stream>>>((const float*)d_in[3], w.s2);

  const float* alpha   = (const float*)d_in[4];
  const float* beta    = (const float*)d_in[5];
  const float* op_W1   = (const float*)d_in[6];
  const float* op_W2   = (const float*)d_in[7];
  const float* op_b1   = (const float*)d_in[8];
  const float* op_b2   = (const float*)d_in[9];
  const float* op_ln_b = (const float*)d_in[10];
  const float* op_ln_g = (const float*)d_in[11];

  SingleP sa  = getSingle(d_in, 12);
  CrossP spa0 = getCross(d_in, 34),  spa1 = getCross(d_in, 67);
  CrossP spaX = getCross(d_in, 100);
  CrossP spe0 = getCross(d_in, 133), spe1 = getCross(d_in, 166);
  CrossP speX = getCross(d_in, 199);

  // depth loop (x1/s1 and x2/s2 branches are independent)
  run_cross(stream, w, spa0, w.x1, w.s1, w.x1);
  run_cross(stream, w, spa1, w.x1, w.s1, w.x1);
  run_cross(stream, w, spe0, w.x2, w.s2, w.x2);
  run_cross(stream, w, spe1, w.x2, w.s2, w.x2);
  // cross fusion
  run_cross(stream, w, spaX, w.x1, w.x2, w.bspa);
  run_cross(stream, w, speX, w.x2, w.x1, w.bspe);
  k_fuse<<<4096, 256, 0, stream>>>(alpha, beta, w.bspa, w.bspe, w.x1);  // fusion -> x1

  // self-attn single block (residual)
  k_ln<0><<<2048, 256, 0, stream>>>(w.x1, 64, sa.n_g, sa.n_b,
                                    (float*)nullptr, 0, w.cat, 128, 0);
  run_mamba(stream, w, sa.m, false, w.cat, 128, (const __bf16*)nullptr, 0, w.mo);
  k_ln<0><<<2048, 256, 0, stream>>>(w.mo, 64, sa.pn_g, sa.pn_b, w.mo, 64,
                                    (__bf16*)nullptr, 0, 0);
  k_add<<<4096, 256, 0, stream>>>(w.mo, w.x1);

  // output projection: LN -> W1+gelu -> W2
  k_ln<0><<<2048, 256, 0, stream>>>(w.x1, 64, op_ln_g, op_ln_b,
                                    (float*)nullptr, 0, w.ybf, 64, 0);
  w2bf(stream, op_W1, 128, 64, 128, 64, w.w_in);
  w2bf(stream, op_W2, 64, 128, 64, 128, w.w_cw);
  gemm(stream, 2, w.ybf, 64, w.w_in, 64, op_b1, w.xz, 128, w.cat, 128, 8, 64);
  gemm(stream, 0, w.cat, 128, w.w_cw, 128, op_b2, w.mo, 64, nullptr, 0, 4, 128);
  k_seq_to_img<<<4096, 256, 0, stream>>>(w.mo, (float*)d_out);
}
